// PointNetSetAbstractionMsg_37898791420318
// MI455X (gfx1250) — matrix operand order repaired and checked by reading, not compiled, not run on hardware
//
#include <hip/hip_runtime.h>
#include <hip/hip_bf16.h>

// ---------------------------------------------------------------------------
// PointNet++ Set Abstraction (MSG) for MI455X / gfx1250, wave32 + WMMA f16.
//   B=16, N=4096, S=1024, C_FEAT=64
//   branches: radius {0.1,0.2,0.4}, k {16,32,64}, widths {{32,32,64},{64,64,128},{64,96,128}}
// Pipeline: fps -> (per branch) ball-query+group(f16) -> [GEMM(wmma f16->f32)
//           weights async-staged to LDS, full-cout x 4 col-tiles per wave,
//           BN stats via LDS accumulators] -> BN finalize -> BN+ReLU (f16)
//           -> last layer: BN+ReLU+max -> d_out.
// ---------------------------------------------------------------------------

#define B_ 16
#define N_ 4096
#define S_ 1024
#define CF_ 64
#define BN_EPS 1e-5f

typedef __attribute__((ext_vector_type(16))) _Float16 v16h;
typedef __attribute__((ext_vector_type(8)))  _Float16 v8h;
typedef __attribute__((ext_vector_type(8)))  float    v8f;

// ---- workspace layout (bytes) ----
#define FPS_IDX_OFF   ((size_t)0)                       // B*S ints        = 64 KB
#define NEWXYZ_OFF    ((size_t)(64*1024))               // B*S*3 f32       = 192 KB
#define STATS_OFF     ((size_t)(512*1024))              // 512 f32 (sum/sumsq/scale/shift)
#define WP_OFF        ((size_t)(768*1024))               // 128*96 f16 packed weights
#define XA_OFF        ((size_t)(1<<20))                 // activations ping (<=192 MB)
#define XB_OFF        (XA_OFF + (size_t)201326592)      // activations pong (<=256 MB)

// ===========================================================================
// 1) Farthest-point sampling: one block per batch, serial over S samples.
// ===========================================================================
__global__ void fps_kernel(const float* __restrict__ xyz, int* __restrict__ fpsIdx,
                           float* __restrict__ newxyz, float* __restrict__ out)
{
    __shared__ float sDist[N_];
    __shared__ float sBestV[256];
    __shared__ int   sBestI[256];
    __shared__ int   sFar;
    const int b = blockIdx.x, tid = threadIdx.x;
    const float* px = xyz + (size_t)b * 3 * N_;   // layout [3][N]
    for (int n = tid; n < N_; n += 256) sDist[n] = 1e10f;
    if (tid == 0) sFar = 0;
    __syncthreads();

    for (int i = 0; i < S_; ++i) {
        const int far = sFar;
        const float cx = px[far], cy = px[N_ + far], cz = px[2 * N_ + far];
        if (tid < 3) {
            float v = px[tid * N_ + far];
            out[(size_t)b * 3 * S_ + tid * S_ + i] = v;           // new_xyz (B,3,S)
            newxyz[((size_t)b * S_ + i) * 3 + tid] = v;           // ws copy [B][S][3]
        }
        if (tid == 0) fpsIdx[b * S_ + i] = far;

        float bv = -1.0f; int bi = 0;
        for (int n = tid; n < N_; n += 256) {
            float dx = px[n] - cx, dy = px[N_ + n] - cy, dz = px[2 * N_ + n] - cz;
            float d  = dx * dx + dy * dy + dz * dz;
            float nd = fminf(sDist[n], d);
            sDist[n] = nd;
            if (nd > bv) { bv = nd; bi = n; }   // ascending n + strict > => first max
        }
        sBestV[tid] = bv; sBestI[tid] = bi;
        __syncthreads();
        for (int off = 128; off > 0; off >>= 1) {
            if (tid < off) {
                float v2 = sBestV[tid + off]; int i2 = sBestI[tid + off];
                if (v2 > sBestV[tid] || (v2 == sBestV[tid] && i2 < sBestI[tid])) {
                    sBestV[tid] = v2; sBestI[tid] = i2;
                }
            }
            __syncthreads();
        }
        if (tid == 0) sFar = sBestI[0];
        __syncthreads();
    }
}

// ===========================================================================
// 2) Ball query + grouping. One wave per centroid (8 centroids / 256-thr block).
//    Writes f16 activation rows X[m][cinPad], m = (b*S+s)*k + slot.
// ===========================================================================
__device__ __forceinline__ void write_row(_Float16* __restrict__ X, size_t rowOff,
                                          const float* __restrict__ fbase,
                                          const float* __restrict__ sx, int n,
                                          float cx, float cy, float cz, int cinPad)
{
    _Float16* row = X + rowOff;
#pragma unroll 8
    for (int c = 0; c < CF_; ++c) row[c] = (_Float16)fbase[(size_t)c * N_ + n];
    row[64] = (_Float16)(sx[n * 3 + 0] - cx);
    row[65] = (_Float16)(sx[n * 3 + 1] - cy);
    row[66] = (_Float16)(sx[n * 3 + 2] - cz);
    for (int c = 67; c < cinPad; ++c) row[c] = (_Float16)0.0f;
}

__global__ void group_kernel(const float* __restrict__ xyz, const float* __restrict__ points,
                             const float* __restrict__ newxyz, _Float16* __restrict__ X,
                             float r2, int k, int cinPad)
{
    __shared__ float sx[N_ * 3];                 // 48 KB staged xyz
    const int bpb = S_ / 8;                      // blocks per batch
    const int b  = blockIdx.x / bpb;
    const int s0 = (blockIdx.x % bpb) * 8;
    for (int i = threadIdx.x; i < N_ * 3; i += blockDim.x) {
        int d = i / N_, n = i % N_;
        sx[n * 3 + d] = xyz[(size_t)b * 3 * N_ + d * N_ + n];
    }
    __syncthreads();

    const int wave = threadIdx.x >> 5, lane = threadIdx.x & 31;
    const int s = s0 + wave;
    const float cx = newxyz[((size_t)b * S_ + s) * 3 + 0];
    const float cy = newxyz[((size_t)b * S_ + s) * 3 + 1];
    const float cz = newxyz[((size_t)b * S_ + s) * 3 + 2];
    const float* fbase = points + (size_t)b * CF_ * N_;
    const size_t mBase = ((size_t)(b * S_ + s)) * (size_t)k;

    int cnt = 0, n0 = 0;
    for (int base = 0; base < N_ && cnt < k; base += 32) {
        const int n = base + lane;
        float dx = sx[n * 3] - cx, dy = sx[n * 3 + 1] - cy, dz = sx[n * 3 + 2] - cz;
        bool in = (dx * dx + dy * dy + dz * dz) <= r2;
        unsigned mask = (unsigned)__ballot(in);
        int slot = cnt + __popc(mask & ((1u << lane) - 1u));
        if (cnt == 0 && mask) n0 = base + __ffs(mask) - 1;   // first point in ball
        if (in && slot < k)
            write_row(X, (mBase + slot) * (size_t)cinPad, fbase, sx, n, cx, cy, cz, cinPad);
        cnt += __popc(mask);
    }
    for (int slot = cnt + lane; slot < k; slot += 32)        // replicate first hit
        write_row(X, (mBase + slot) * (size_t)cinPad, fbase, sx, n0, cx, cy, cz, cinPad);
}

// ===========================================================================
// 3) GEMM: Y[m][cout] = W[cout x cinPad] * X[m][cinPad]^T + bias, pre-BN f16.
//    - Weight panel async-copied to LDS once per block (ASYNCcnt path).
//    - One wave owns CT=4 column tiles and ALL RT row tiles: each B tile is
//      loaded once (v16h = 32B/lane) and reused for RT WMMAs (reuse_b hint).
//    - BN sum/sumsq accumulated in LDS (ds_add_f32), flushed once per block.
// ===========================================================================
__device__ __forceinline__ v16h load_a(const _Float16* ap)
{
    v8h lo = *(const v8h*)ap;            // ds_load_b128
    v8h hi = *(const v8h*)(ap + 8);
    v16h a;
#pragma unroll
    for (int i = 0; i < 8; ++i) { a[i] = lo[i]; a[i + 8] = hi[i]; }
    return a;
}

template<int RT, int KT>
__global__ void gemm_bn_stats_kernel(const _Float16* __restrict__ Wp,
                                     const _Float16* __restrict__ Xin,
                                     const float* __restrict__ bias,
                                     _Float16* __restrict__ Yout,
                                     float* __restrict__ gsum, float* __restrict__ gsumsq,
                                     int M)
{
    constexpr int cinPad = KT * 32;
    constexpr int cout   = RT * 16;
    constexpr int CT     = 4;                     // column tiles per wave
    __shared__ _Float16 sW[cout * cinPad];        // <= 128*96*2 = 24 KB
    __shared__ float    sStat[2 * cout];          // block-local BN sum / sumsq

    if (threadIdx.x < 2 * cout) sStat[threadIdx.x] = 0.0f;

    // --- async-stage the weight panel into LDS (16B chunks per lane) ---
    constexpr int nChunks = (cout * cinPad * 2) / 16;
    for (int i = threadIdx.x; i < nChunks; i += 256) {
        unsigned ldsOff = (unsigned)(uintptr_t)(const void*)((const char*)sW + i * 16);
        const void* g   = (const char*)Wp + i * 16;
        asm volatile("global_load_async_to_lds_b128 %0, %1, off"
                     :: "v"(ldsOff), "v"(g) : "memory");
    }
    asm volatile("s_wait_asynccnt 0x0" ::: "memory");
    __syncthreads();

    const int lane  = threadIdx.x & 31;
    const int wave  = threadIdx.x >> 5;
    const int l15   = lane & 15;
    const int hiK8  = (lane & 16) ? 16 : 0;       // A: K halves {0..15}/{16..31}
    const int hiK16 = (lane & 16) ? 16 : 0;       // B: K halves {0..15}/{16..31}
    const int rHalf = (lane & 16) ? 8 : 0;        // D: rows base / base+8

#pragma unroll 1
    for (int t = 0; t < CT; ++t) {
        const int ct = (blockIdx.x * 8 + wave) * CT + t;    // column tile (m / 16)
        const _Float16* bBase = Xin + (size_t)(ct * 16 + l15) * cinPad + hiK16;
        if (t + 1 < CT)
            __builtin_prefetch(bBase + 16 * cinPad, 0, 1);  // global_prefetch next tile

        v8f acc[RT];
#pragma unroll
        for (int rt = 0; rt < RT; ++rt) acc[rt] = (v8f){};

#pragma unroll
        for (int kt = 0; kt < KT; ++kt) {
            const v16h bmat = *(const v16h*)(bBase + kt * 32);
            // rt = 0: fresh B operand
            acc[0] = __builtin_amdgcn_wmma_f32_16x16x32_f16(
                false, load_a(sW + (0 * 16 + l15) * cinPad + hiK8 + kt * 32),
                false, bmat, (short)0, acc[0], false, false);
            // rt >= 1: identical B operand -> reuse_b hint
#pragma unroll
            for (int rt = 1; rt < RT; ++rt) {
                acc[rt] = __builtin_amdgcn_wmma_f32_16x16x32_f16(
                    false, load_a(sW + (rt * 16 + l15) * cinPad + hiK8 + kt * 32),
                    false, bmat, (short)0, acc[rt], false, true);
            }
        }

        const int m = ct * 16 + l15;
#pragma unroll
        for (int rt = 0; rt < RT; ++rt) {
            const int rbase = rt * 16 + rHalf;
            float vals[8];
#pragma unroll
            for (int i = 0; i < 8; ++i) vals[i] = acc[rt][i] + bias[rbase + i];

            v8h st;
#pragma unroll
            for (int i = 0; i < 8; ++i) st[i] = (_Float16)vals[i];
            *(v8h*)(Yout + (size_t)m * cout + rbase) = st;

#pragma unroll
            for (int i = 0; i < 8; ++i) {
                atomicAdd(&sStat[rbase + i], vals[i]);                 // ds_add_f32
                atomicAdd(&sStat[cout + rbase + i], vals[i] * vals[i]);
            }
        }
    }

    __syncthreads();
    if (threadIdx.x < cout)
        atomicAdd(&gsum[threadIdx.x], sStat[threadIdx.x]);
    else if (threadIdx.x < 2 * cout)
        atomicAdd(&gsumsq[threadIdx.x - cout], sStat[threadIdx.x]);
}

// ===========================================================================
// 4) Small helpers.
// ===========================================================================
__global__ void pack_w_kernel(const float* __restrict__ W, _Float16* __restrict__ Wp,
                              int cout, int cin, int cinPad)
{
    int t = blockIdx.x * blockDim.x + threadIdx.x;
    if (t >= cout * cinPad) return;
    int r = t / cinPad, c = t % cinPad;
    Wp[t] = (c < cin) ? (_Float16)W[r * cin + c] : (_Float16)0.0f;
}

__global__ void zero_stats_kernel(float* __restrict__ stat)
{
    if (threadIdx.x < 256) stat[threadIdx.x] = 0.0f;   // sum[128] + sumsq[128]
}

__global__ void bn_finalize_kernel(const float* __restrict__ sum, const float* __restrict__ sumsq,
                                   const float* __restrict__ gamma, const float* __restrict__ beta,
                                   float* __restrict__ scale, float* __restrict__ shift,
                                   float invCount, int cout)
{
    int c = threadIdx.x;
    if (c >= cout) return;
    float mu  = sum[c] * invCount;
    float var = sumsq[c] * invCount - mu * mu;
    float sc  = gamma[c] * rsqrtf(var + BN_EPS);
    scale[c] = sc;
    shift[c] = beta[c] - mu * sc;
}

__global__ void bn_relu_kernel(_Float16* __restrict__ Y, const float* __restrict__ scale,
                               const float* __restrict__ shift, int cout, size_t tot)
{
    size_t t = (size_t)blockIdx.x * blockDim.x + threadIdx.x;
    if (t >= tot) return;
    int c = (int)(t % (size_t)cout);
    float v = (float)Y[t] * scale[c] + shift[c];
    Y[t] = (_Float16)fmaxf(v, 0.0f);
}

__global__ void bn_relu_max_kernel(const _Float16* __restrict__ Y,
                                   const float* __restrict__ scale, const float* __restrict__ shift,
                                   float* __restrict__ out, int k, int cout, int coff)
{
    int t = blockIdx.x * blockDim.x + threadIdx.x;
    if (t >= B_ * S_ * cout) return;
    int c = t % cout;
    int s = (t / cout) % S_;
    int b = t / (cout * S_);
    const size_t mb = ((size_t)(b * S_ + s)) * (size_t)k;
    const float sc = scale[c], sh = shift[c];
    float mx = 0.0f;                                   // relu output >= 0
    for (int kk = 0; kk < k; ++kk) {
        float v = (float)Y[(mb + kk) * (size_t)cout + c] * sc + sh;
        mx = fmaxf(mx, fmaxf(v, 0.0f));
    }
    out[(size_t)B_ * 3 * S_ + (size_t)b * 320 * S_ + (size_t)(coff + c) * S_ + s] = mx;
}

// ===========================================================================
extern "C" void kernel_launch(void* const* d_in, const int* in_sizes, int n_in,
                              void* d_out, int out_size, void* d_ws, size_t ws_size,
                              hipStream_t stream)
{
    (void)in_sizes; (void)n_in; (void)out_size; (void)ws_size;
    const float* xyz    = (const float*)d_in[0];
    const float* points = (const float*)d_in[1];
    char*  ws  = (char*)d_ws;
    float* out = (float*)d_out;

    int*      fpsIdx = (int*)(ws + FPS_IDX_OFF);
    float*    newxyz = (float*)(ws + NEWXYZ_OFF);
    float*    stat   = (float*)(ws + STATS_OFF);
    float*    sum    = stat, *sumsq = stat + 128, *scale = stat + 256, *shift = stat + 384;
    _Float16* Wp     = (_Float16*)(ws + WP_OFF);
    _Float16* XA     = (_Float16*)(ws + XA_OFF);
    _Float16* XB     = (_Float16*)(ws + XB_OFF);

    fps_kernel<<<B_, 256, 0, stream>>>(xyz, fpsIdx, newxyz, out);

    const int   widths[3][3] = {{32, 32, 64}, {64, 64, 128}, {64, 96, 128}};
    const int   ksamp[3]     = {16, 32, 64};
    const float rad2[3]      = {0.01f, 0.04f, 0.16f};

    int coff = 0;
    for (int br = 0; br < 3; ++br) {
        const int k = ksamp[br];
        const int M = B_ * S_ * k;                    // 262144 / 524288 / 1048576

        group_kernel<<<B_ * S_ / 8, 256, 0, stream>>>(xyz, points, newxyz, XA,
                                                      rad2[br], k, 96);

        _Float16* cur = XA;
        _Float16* nxt = XB;
        int cin = 67, cinPad = 96;
        for (int l = 0; l < 3; ++l) {
            const int cout = widths[br][l];
            const int pbase = 2 + br * 12 + l * 4;
            const float* W     = (const float*)d_in[pbase + 0];
            const float* bias  = (const float*)d_in[pbase + 1];
            const float* gamma = (const float*)d_in[pbase + 2];
            const float* beta  = (const float*)d_in[pbase + 3];

            pack_w_kernel<<<(cout * cinPad + 255) / 256, 256, 0, stream>>>(W, Wp, cout, cin, cinPad);
            zero_stats_kernel<<<1, 256, 0, stream>>>(sum);

            dim3 gg(M / 512);                         // 8 waves/block, 4 col-tiles/wave
            const int rtv = cout / 16, ktv = cinPad / 32;
            if      (rtv == 2 && ktv == 1) gemm_bn_stats_kernel<2,1><<<gg,256,0,stream>>>(Wp,cur,bias,nxt,sum,sumsq,M);
            else if (rtv == 2 && ktv == 3) gemm_bn_stats_kernel<2,3><<<gg,256,0,stream>>>(Wp,cur,bias,nxt,sum,sumsq,M);
            else if (rtv == 4 && ktv == 1) gemm_bn_stats_kernel<4,1><<<gg,256,0,stream>>>(Wp,cur,bias,nxt,sum,sumsq,M);
            else if (rtv == 4 && ktv == 2) gemm_bn_stats_kernel<4,2><<<gg,256,0,stream>>>(Wp,cur,bias,nxt,sum,sumsq,M);
            else if (rtv == 4 && ktv == 3) gemm_bn_stats_kernel<4,3><<<gg,256,0,stream>>>(Wp,cur,bias,nxt,sum,sumsq,M);
            else if (rtv == 6 && ktv == 2) gemm_bn_stats_kernel<6,2><<<gg,256,0,stream>>>(Wp,cur,bias,nxt,sum,sumsq,M);
            else if (rtv == 8 && ktv == 2) gemm_bn_stats_kernel<8,2><<<gg,256,0,stream>>>(Wp,cur,bias,nxt,sum,sumsq,M);
            else if (rtv == 8 && ktv == 3) gemm_bn_stats_kernel<8,3><<<gg,256,0,stream>>>(Wp,cur,bias,nxt,sum,sumsq,M);

            bn_finalize_kernel<<<1, cout, 0, stream>>>(sum, sumsq, gamma, beta,
                                                       scale, shift, 1.0f / (float)M, cout);
            if (l < 2) {
                size_t tot = (size_t)M * cout;
                bn_relu_kernel<<<(unsigned)((tot + 255) / 256), 256, 0, stream>>>(nxt, scale, shift, cout, tot);
            } else {
                int tot = B_ * S_ * cout;
                bn_relu_max_kernel<<<(tot + 255) / 256, 256, 0, stream>>>(nxt, scale, shift,
                                                                          out, k, cout, coff);
            }
            _Float16* tmp = cur; cur = nxt; nxt = tmp;
            cin = cout; cinPad = cout;                // all layer widths are multiples of 32
        }
        coff += widths[br][2];
    }
}
